// RotaryCrossAttention_49400713839104
// MI455X (gfx1250) — compile-verified, matrix operand order checked by reading, not run on hardware
//
#include <hip/hip_runtime.h>
#include <hip/hip_fp16.h>

typedef __attribute__((ext_vector_type(16))) _Float16 v16h;
typedef __attribute__((ext_vector_type(4)))  _Float16 v4h;
typedef __attribute__((ext_vector_type(8)))  float    v8f;

union FragU {
  v16h h;
  uint4 q[2];
};

// Load a 16x32 f16 A/B operand fragment for V_WMMA_F32_16X16X32_F16.
// Per ISA 7.12.2: lanes 0-15 hold K=0..7 (vgpr0-3) and K=16..23 (vgpr4-7);
// lanes 16-31 hold K=8..15 and K=24..31 -> two 16-byte chunks at
// byte offsets 16*half and 32 + 16*half from the row base.
__device__ __forceinline__ v16h ld_frag(const _Float16* row, int half_id) {
  FragU f;
  f.q[0] = *(const uint4*)(row + 8 * half_id);
  f.q[1] = *(const uint4*)(row + 16 + 8 * half_id);
  return f.h;
}

__device__ __forceinline__ v8f wmma_f16(v16h a, v16h b, v8f c) {
  return __builtin_amdgcn_wmma_f32_16x16x32_f16(
      /*neg_a=*/false, a, /*neg_b=*/false, b,
      /*c_mod=*/(short)0, c, /*reuse_a=*/false, /*reuse_b=*/false);
}

// CDNA5 async DMA: copy 16 bytes global -> LDS, tracked by ASYNCcnt.
// lds_off is the byte offset within the workgroup LDS allocation
// (low 32 bits of the flat shared address).
__device__ __forceinline__ void async_copy16(unsigned lds_off, const void* gaddr) {
  asm volatile("global_load_async_to_lds_b128 %0, %1, off"
               :
               : "v"(lds_off), "v"(gaddr)
               : "memory");
}

__device__ __forceinline__ void wait_async0() {
  asm volatile("s_wait_asynccnt 0x0" ::: "memory");
}

__device__ __forceinline__ unsigned lds_off_of(const void* p) {
  return (unsigned)(uintptr_t)p;
}

// ---------------------------------------------------------------------------
// float -> f16 conversion pass (vectorized, bandwidth-trivial)
// ---------------------------------------------------------------------------
__global__ __launch_bounds__(256) void cvt_f2h(const float* __restrict__ src,
                                               _Float16* __restrict__ dst, int n4) {
  int i = blockIdx.x * blockDim.x + threadIdx.x;
  if (i < n4) {
    float4 v = ((const float4*)src)[i];
    v4h h = {(_Float16)v.x, (_Float16)v.y, (_Float16)v.z, (_Float16)v.w};
    ((v4h*)dst)[i] = h;
  }
}

// ---------------------------------------------------------------------------
// GEMM: C[M,N] = A[M,K] @ W[N,K]^T   (f16 inputs, f32 accum)
// Block tile 64x64, 128 threads = 4 waves; k-step 64; double-buffered LDS
// filled by async DMA.  Wave w owns rows w*16..w*16+15 across all 64 cols.
// ---------------------------------------------------------------------------
template <bool OUT_HALF>
__global__ __launch_bounds__(128) void gemm_h(const _Float16* __restrict__ A,
                                              const _Float16* __restrict__ W,
                                              void* __restrict__ Cout,
                                              int M, int N, int K) {
  __shared__ __align__(16) _Float16 sA[2][64 * 64];
  __shared__ __align__(16) _Float16 sW[2][64 * 64];
  const int tid   = threadIdx.x;
  const int wave  = tid >> 5;
  const int lane  = tid & 31;
  const int lhalf = lane >> 4;
  const int l16   = lane & 15;
  const int m0 = blockIdx.x * 64;
  const int n0 = blockIdx.y * 64;
  const unsigned sA_base = lds_off_of(&sA[0][0]);
  const unsigned sW_base = lds_off_of(&sW[0][0]);

  auto issue_tiles = [&](int buf, int kb) {
    unsigned ab = sA_base + (unsigned)buf * 8192;
    unsigned wb = sW_base + (unsigned)buf * 8192;
#pragma unroll
    for (int i = 0; i < 4; ++i) {
      int idx = tid + 128 * i;          // 0..511 over 64 rows x 8 chunks
      int r = idx >> 3, c = idx & 7;
      async_copy16(ab + (unsigned)idx * 16, A + (size_t)(m0 + r) * K + kb + c * 8);
      async_copy16(wb + (unsigned)idx * 16, W + (size_t)(n0 + r) * K + kb + c * 8);
    }
  };

  v8f zero = {};
  v8f acc[4] = {zero, zero, zero, zero};

  const int nk = K >> 6;                // k-steps of 64
  issue_tiles(0, 0);
  for (int t = 0; t < nk; ++t) {
    wait_async0();
    __syncthreads();                    // tile t resident; everyone done with buf t&1
    if (t + 1 < nk) issue_tiles((t + 1) & 1, (t + 1) << 6);
    const _Float16* pa = &sA[t & 1][(wave * 16 + l16) * 64];
    const _Float16* pb0 = &sW[t & 1][l16 * 64];
#pragma unroll
    for (int ks = 0; ks < 2; ++ks) {
      v16h a = ld_frag(pa + ks * 32, lhalf);
#pragma unroll
      for (int nt = 0; nt < 4; ++nt) {
        v16h b = ld_frag(pb0 + nt * 16 * 64 + ks * 32, lhalf);
        acc[nt] = wmma_f16(a, b, acc[nt]);
      }
    }
    __syncthreads();                    // compute t done before buf t&1 is refilled
  }

#pragma unroll
  for (int nt = 0; nt < 4; ++nt) {
#pragma unroll
    for (int r = 0; r < 8; ++r) {
      int m = m0 + wave * 16 + r + 8 * lhalf;
      int n = n0 + nt * 16 + l16;
      float v = acc[nt][r];
      if (OUT_HALF)
        ((_Float16*)Cout)[(size_t)m * N + n] = (_Float16)v;
      else
        ((float*)Cout)[(size_t)m * N + n] = v;
    }
  }
}

// ---------------------------------------------------------------------------
// RoPE (interleaved): out[2i] = x[2i]*cos - x[2i+1]*sin ; out[2i+1] = x[2i+1]*cos + x[2i]*sin
// theta_i = 10000^(i/32), ang = pos/theta_i  (HEAD_DIM = 64)
// ---------------------------------------------------------------------------
__global__ void rope_q_kernel(const _Float16* __restrict__ qp,   // [B,LQ,H,64]
                              _Float16* __restrict__ qh) {        // [B,H,LQ,64]
  int idx = blockIdx.x * blockDim.x + threadIdx.x;  // B*LQ*H*32
  int i  = idx & 31;
  int h  = (idx >> 5) & 15;
  int lq = (idx >> 9) & 1023;
  int b  = idx >> 19;
  const _Float16* src = qp + ((((size_t)b * 1024 + lq) * 16 + h) * 64 + 2 * i);
  float x0 = (float)src[0], x1 = (float)src[1];
  float theta = __powf(10000.f, (float)i * (1.f / 32.f));
  float ang = (float)lq / theta;
  float s, c;
  __sincosf(ang, &s, &c);
  _Float16* dst = qh + ((((size_t)b * 16 + h) * 1024 + lq) * 64 + 2 * i);
  dst[0] = (_Float16)(x0 * c - x1 * s);
  dst[1] = (_Float16)(x1 * c + x0 * s);
}

__global__ void rope_kv_kernel(const _Float16* __restrict__ kvp,  // [B,LK,H,128]
                               _Float16* __restrict__ kh,          // [B,H,LK,64]
                               _Float16* __restrict__ vt) {        // [B,H,64,LK]
  int idx = blockIdx.x * blockDim.x + threadIdx.x;  // B*LK*H*32
  int i  = idx & 31;
  int h  = (idx >> 5) & 15;
  int lk = (idx >> 9) & 2047;
  int b  = idx >> 20;
  const _Float16* src = kvp + (((size_t)b * 2048 + lk) * 16 + h) * 128;
  float x0 = (float)src[2 * i], x1 = (float)src[2 * i + 1];
  float theta = __powf(10000.f, (float)i * (1.f / 32.f));
  float ang = (float)lk / theta;
  float s, c;
  __sincosf(ang, &s, &c);
  _Float16* kdst = kh + ((((size_t)b * 16 + h) * 2048 + lk) * 64 + 2 * i);
  kdst[0] = (_Float16)(x0 * c - x1 * s);
  kdst[1] = (_Float16)(x1 * c + x0 * s);
  size_t vb = ((size_t)b * 16 + h) * 64;
  vt[(vb + 2 * i)     * 2048 + lk] = src[64 + 2 * i];
  vt[(vb + 2 * i + 1) * 2048 + lk] = src[64 + 2 * i + 1];
}

// ---------------------------------------------------------------------------
// Flash attention: one block = (b,h, 64-row q tile). 128 threads = 4 waves,
// wave w owns q rows w*16..w*16+15.  64-key blocks, double-buffered async
// K/V tiles, online softmax.  Output f16 [b,lq,h,d].
// ---------------------------------------------------------------------------
__global__ __launch_bounds__(128) void attn_kernel(const _Float16* __restrict__ qh,
                                                   const _Float16* __restrict__ kh,
                                                   const _Float16* __restrict__ vt,
                                                   _Float16* __restrict__ attnh) {
  const int LQ = 1024, LK = 2048, H = 16, D = 64;
  __shared__ __align__(16) _Float16 sK[2][64 * 64];   // [key][d]
  __shared__ __align__(16) _Float16 sV[2][64 * 64];   // [d][key]
  __shared__ __align__(16) _Float16 sP[4][16 * 64];   // per-wave P tile

  const int tid   = threadIdx.x;
  const int wave  = tid >> 5;
  const int lane  = tid & 31;
  const int lhalf = lane >> 4;
  const int l16   = lane & 15;
  const int qt = blockIdx.x & 15;   // LQ/64 = 16 tiles
  const int bh = blockIdx.x >> 4;   // b*16 + h
  const unsigned sK_base = lds_off_of(&sK[0][0]);
  const unsigned sV_base = lds_off_of(&sV[0][0]);

  auto issue_kv = [&](int buf, int j) {
    const _Float16* gk = kh + (size_t)bh * LK * 64 + (size_t)j * 4096;  // contiguous 8KB
    unsigned kb = sK_base + (unsigned)buf * 8192;
    unsigned vb = sV_base + (unsigned)buf * 8192;
#pragma unroll
    for (int i = 0; i < 4; ++i) {
      int idx = tid + 128 * i;          // 0..511
      async_copy16(kb + (unsigned)idx * 16, gk + idx * 8);
      int r = idx >> 3, c = idx & 7;
      async_copy16(vb + (unsigned)idx * 16,
                   vt + ((size_t)bh * 64 + r) * LK + j * 64 + c * 8);
    }
  };

  // Q fragments for this wave's 16 rows (two k-chunks of 32), from global
  v16h qa[2];
  {
    const _Float16* qrow = qh + ((size_t)bh * LQ + qt * 64 + wave * 16 + l16) * 64;
    qa[0] = ld_frag(qrow, lhalf);
    qa[1] = ld_frag(qrow + 32, lhalf);
  }

  v8f zero = {};
  v8f acc_o[4] = {zero, zero, zero, zero};
  float m_i[8], l_i[8];
#pragma unroll
  for (int r = 0; r < 8; ++r) { m_i[r] = -1e30f; l_i[r] = 0.f; }

  const int NJ = LK / 64;
  issue_kv(0, 0);
  for (int j = 0; j < NJ; ++j) {
    wait_async0();
    __syncthreads();                    // tile j resident; buf j&1 consumers of j-2 done
    if (j + 1 < NJ) issue_kv((j + 1) & 1, j + 1);
    const _Float16* cK = &sK[j & 1][0];
    const _Float16* cV = &sV[j & 1][0];

    // S = Q @ K^T  (4 n-tiles x 2 k-steps)
    v8f s[4];
#pragma unroll
    for (int nt = 0; nt < 4; ++nt) {
      v8f c = zero;
#pragma unroll
      for (int ks = 0; ks < 2; ++ks) {
        v16h b = ld_frag(cK + (nt * 16 + l16) * 64 + ks * 32, lhalf);
        c = wmma_f16(qa[ks], b, c);
      }
      s[nt] = c;
    }

    // Online softmax. C-layout: vgpr r <-> row (r + 8*lhalf); width-16 shuffles.
    float scale_r[8];
#pragma unroll
    for (int r = 0; r < 8; ++r) {
      float v = fmaxf(fmaxf(s[0][r], s[1][r]), fmaxf(s[2][r], s[3][r]));
#pragma unroll
      for (int off = 8; off >= 1; off >>= 1) v = fmaxf(v, __shfl_xor(v, off, 16));
      float mnew = fmaxf(m_i[r], v);
      scale_r[r] = __expf(m_i[r] - mnew);
      m_i[r] = mnew;
    }

    float rsum[8];
#pragma unroll
    for (int r = 0; r < 8; ++r) rsum[r] = 0.f;
    _Float16* pw = &sP[wave][0];
#pragma unroll
    for (int nt = 0; nt < 4; ++nt) {
#pragma unroll
      for (int r = 0; r < 8; ++r) {
        float p = __expf(s[nt][r] - m_i[r]);
        rsum[r] += p;
        pw[(r + 8 * lhalf) * 64 + nt * 16 + l16] = (_Float16)p;
      }
    }
#pragma unroll
    for (int r = 0; r < 8; ++r) {
      float v = rsum[r];
#pragma unroll
      for (int off = 8; off >= 1; off >>= 1) v += __shfl_xor(v, off, 16);
      l_i[r] = l_i[r] * scale_r[r] + v;
    }
#pragma unroll
    for (int nt = 0; nt < 4; ++nt)
#pragma unroll
      for (int r = 0; r < 8; ++r) acc_o[nt][r] *= scale_r[r];

    // O += P @ V : re-fetch P from per-wave LDS in A-operand layout
    v16h pa[2];
    pa[0] = ld_frag(&pw[l16 * 64], lhalf);
    pa[1] = ld_frag(&pw[l16 * 64 + 32], lhalf);
#pragma unroll
    for (int nt = 0; nt < 4; ++nt) {
      v8f c = acc_o[nt];
#pragma unroll
      for (int ks = 0; ks < 2; ++ks) {
        v16h b = ld_frag(cV + (nt * 16 + l16) * 64 + ks * 32, lhalf);
        c = wmma_f16(pa[ks], b, c);
      }
      acc_o[nt] = c;
    }
    __syncthreads();                    // compute j done before buf j&1 is refilled
  }

  // Normalize and write [b, lq, h, d] f16 (A-input of the Wo GEMM)
  int b = bh >> 4, h = bh & 15;
#pragma unroll
  for (int nt = 0; nt < 4; ++nt) {
#pragma unroll
    for (int r = 0; r < 8; ++r) {
      int lq = qt * 64 + wave * 16 + r + 8 * lhalf;
      int d  = nt * 16 + l16;
      attnh[(((size_t)b * LQ + lq) * H + h) * D + d] =
          (_Float16)(acc_o[nt][r] / l_i[r]);
    }
  }
}

// ---------------------------------------------------------------------------
extern "C" void kernel_launch(void* const* d_in, const int* in_sizes, int n_in,
                              void* d_out, int out_size, void* d_ws, size_t ws_size,
                              hipStream_t stream) {
  (void)in_sizes; (void)n_in; (void)out_size; (void)ws_size;
  const float* x   = (const float*)d_in[0];  // [8,1024,1024]
  const float* y   = (const float*)d_in[1];  // [8,2048,1024]
  const float* Wq  = (const float*)d_in[2];  // [1024,1024]
  const float* Wkv = (const float*)d_in[3];  // [2048,1024]
  const float* Wo  = (const float*)d_in[4];  // [1024,1024]
  float* out = (float*)d_out;                // [8,1024,1024]

  char* ws = (char*)d_ws;
  const size_t MB = 1ull << 20;
  _Float16* xh   = (_Float16*)(ws + 0 * MB);     // 16 MB
  _Float16* yh   = (_Float16*)(ws + 16 * MB);    // 32 MB
  _Float16* wqh  = (_Float16*)(ws + 48 * MB);    //  2 MB
  _Float16* wkvh = (_Float16*)(ws + 50 * MB);    //  4 MB
  _Float16* woh  = (_Float16*)(ws + 54 * MB);    //  2 MB
  _Float16* tmp_h = (_Float16*)(ws + 56 * MB);   // 64 MB (qp / kvp / attn)
  _Float16* qh = (_Float16*)(ws + 120 * MB);     // 16 MB
  _Float16* kh = (_Float16*)(ws + 136 * MB);     // 32 MB
  _Float16* vt = (_Float16*)(ws + 168 * MB);     // 32 MB  (total 200 MB)

  // Convert everything to f16 once (bandwidth-trivial passes)
  cvt_f2h<<<8192, 256, 0, stream>>>(x, xh, 8 * 1024 * 1024 / 4);
  cvt_f2h<<<16384, 256, 0, stream>>>(y, yh, 8 * 2048 * 1024 / 4);
  cvt_f2h<<<1024, 256, 0, stream>>>(Wq, wqh, 1024 * 1024 / 4);
  cvt_f2h<<<2048, 256, 0, stream>>>(Wkv, wkvh, 2048 * 1024 / 4);
  cvt_f2h<<<1024, 256, 0, stream>>>(Wo, woh, 1024 * 1024 / 4);

  dim3 blk(128);
  // Q = x @ Wq^T
  gemm_h<true><<<dim3(8192 / 64, 1024 / 64), blk, 0, stream>>>(xh, wqh, tmp_h, 8192, 1024, 1024);
  rope_q_kernel<<<(8 * 1024 * 16 * 32) / 256, 256, 0, stream>>>(tmp_h, qh);
  // KV = y @ Wkv^T
  gemm_h<true><<<dim3(16384 / 64, 2048 / 64), blk, 0, stream>>>(yh, wkvh, tmp_h, 16384, 2048, 1024);
  rope_kv_kernel<<<(8 * 2048 * 16 * 32) / 256, 256, 0, stream>>>(tmp_h, kh, vt);
  // Attention -> [b,lq,h,d] f16 in tmp
  attn_kernel<<<dim3(2048), blk, 0, stream>>>(qh, kh, vt, tmp_h);
  // out = attn @ Wo^T (f32 out)
  gemm_h<false><<<dim3(8192 / 64, 1024 / 64), blk, 0, stream>>>(tmp_h, woh, out, 8192, 1024, 1024);
}
